// Attention_2765958938679
// MI455X (gfx1250) — matrix operand order repaired and checked by reading, not compiled, not run on hardware
//
#include <hip/hip_runtime.h>
#include <hip/hip_bf16.h>

// ---------------------------------------------------------------------------
// Problem constants (from reference)
// ---------------------------------------------------------------------------
#define BATCH      16
#define NTOK       740      // 16^2 + 22^2
#define CDIM       512
#define NHEAD      16
#define HDIM       32       // CDIM / NHEAD
#define MROWS      (BATCH * NTOK)        // 11840
#define TEMP_L     16
#define TARGET_L   22
#define TEMP_N     256      // 16*16
#define TARGET_N   484      // 22*22
#define SCALE_Q    0.17677669529663687f  // 32^-0.5

// attention kernel tiling
#define QBLK       64                    // query rows per workgroup (4 waves x 16)
#define KPAD       768                   // padded key dim (48 tiles of 16)
// LDS: S panel f32 [64][768] + P panel bf16 [64][768] + rowsum[64]
#define S_FLOATS   (QBLK * KPAD)                          // 49152 floats
#define ATTN_LDS   (S_FLOATS * 4 + QBLK * KPAD * 2 + QBLK * 4)  // 295168 B (<320KB WGP LDS)

typedef __bf16 bf16;
typedef bf16  v16bf __attribute__((ext_vector_type(16)));
typedef float v8f   __attribute__((ext_vector_type(8)));
typedef unsigned int u32x4 __attribute__((ext_vector_type(4)));
typedef int          i32x4 __attribute__((ext_vector_type(4)));
typedef int          i32x8 __attribute__((ext_vector_type(8)));

__device__ inline v8f wmma_bf16f32(const v16bf& a, const v16bf& b, const v8f& c) {
    // D(16x16 f32) = A(16x32 bf16) * B(32x16 bf16) + C
    return __builtin_amdgcn_wmma_f32_16x16x32_bf16(
        /*neg_a=*/false, a, /*neg_b=*/false, b,
        /*c_mod=*/(short)0, c, /*reuse_a=*/false, /*reuse_b=*/false);
}

// load 16 contiguous bf16 (32B) via two 16B chunks (16B alignment required)
__device__ inline v16bf load16bf(const bf16* p) {
    v16bf r;
    ((uint4*)&r)[0] = ((const uint4*)p)[0];
    ((uint4*)&r)[1] = ((const uint4*)p)[1];
    return r;
}
// load 16 bf16 from two separate 16B chunks
__device__ inline v16bf load16bf2(const bf16* lo, const bf16* hi) {
    v16bf r;
    ((uint4*)&r)[0] = *(const uint4*)lo;
    ((uint4*)&r)[1] = *(const uint4*)hi;
    return r;
}
// pack 4 floats -> 4 bf16 and return as uint2 (8-byte store payload)
__device__ inline uint2 pack4bf(float a, float b, float c, float d) {
    union { bf16 h[4]; uint2 u; } pk;
    pk.h[0] = (bf16)a; pk.h[1] = (bf16)b; pk.h[2] = (bf16)c; pk.h[3] = (bf16)d;
    return pk.u;
}

// ---------------------------------------------------------------------------
// TDM: async 2D tile load (bf16) global -> LDS. Issued from ONE wave only
// (TDM ignores EXEC; gate at wave granularity). Descriptor per ISA D# layout.
//   tile: tile_d1 rows x tile_d0 elements, row stride `stride0` elements.
//   rows_avail: rows valid from gaddr (hardware zero-fills past tensor_dim1).
// This toolchain exposes the 6-arg builtin: (g0, g1, g2, g3, g4, cpol).
// ---------------------------------------------------------------------------
__device__ inline void tdm_load_tile_bf16(unsigned int lds_addr, const bf16* gptr,
                                          unsigned int tile_d0, unsigned int tile_d1,
                                          unsigned int stride0, unsigned int rows_avail)
{
    const unsigned long long ga = (unsigned long long)(uintptr_t)gptr;
    u32x4 g0;
    g0[0] = 1u;                                   // count=1, user descriptor
    g0[1] = lds_addr;                             // LDS byte offset
    g0[2] = (unsigned int)ga;                     // global_addr[31:0]
    g0[3] = (unsigned int)((ga >> 32) & 0x1FFFFFFu) | 0x80000000u;  // addr[56:32] | type=2
    i32x8 g1;
    g1[0] = 0x00010000;                           // workgroup_mask=0, data_size=1 (2B)
    g1[1] = (int)((stride0 & 0xFFFFu) << 16);     // tensor_dim0[15:0] (=row length)
    g1[2] = (int)(((stride0 >> 16) & 0xFFFFu) |   // tensor_dim0[31:16]
                  ((rows_avail & 0xFFFFu) << 16));// tensor_dim1[15:0]
    g1[3] = (int)(((rows_avail >> 16) & 0xFFFFu) |// tensor_dim1[31:16]
                  ((tile_d0 & 0xFFFFu) << 16));   // tile_dim0
    g1[4] = (int)(tile_d1 & 0xFFFFu);             // tile_dim1 (tile_dim2=0)
    g1[5] = (int)stride0;                         // tensor_dim0_stride[31:0]
    g1[6] = 0;                                    // stride0[47:32]=0, dim1_stride[15:0]=0
    g1[7] = 0;
    const i32x4 z4 = {0, 0, 0, 0};                // groups 2/3 unused (2D tensor)
    const i32x8 z8 = {0, 0, 0, 0, 0, 0, 0, 0};
    __builtin_amdgcn_tensor_load_to_lds(g0, g1, z4, z4, z8, 0);
}

// ---------------------------------------------------------------------------
// Kernel 1: qkv = x @ W_qkv + b_qkv, scattered to q/k [B,H,N,hd] bf16 and
// v transposed-padded [B,H,hd,KPAD] bf16. 256 threads = 8 waves,
// tile 128(M) x 64(N), K-step 32 (WMMA K).
// ---------------------------------------------------------------------------
__global__ __launch_bounds__(256)
void qkv_kernel(const float* __restrict__ x,      // [MROWS][512]
                const float* __restrict__ Wqkv,   // [512][1536]
                const float* __restrict__ bqkv,   // [1536]
                bf16* __restrict__ qws, bf16* __restrict__ kws,
                bf16* __restrict__ vtws)          // [B,H,32,KPAD]
{
    __shared__ __align__(16) bf16 As[128][32];
    __shared__ __align__(16) bf16 BsT[64][32];    // transposed: [ncol][k]

    const int tid  = threadIdx.x;
    const int wave = tid >> 5;
    const int lane = tid & 31;
    const int half = lane >> 4;
    const int l16  = lane & 15;
    const int mbase = blockIdx.x * 128;
    const int nbase = blockIdx.y * 64;

    v8f acc[4];
    for (int nt = 0; nt < 4; ++nt)
        for (int j = 0; j < 8; ++j) acc[nt][j] = 0.0f;

    for (int k0 = 0; k0 < CDIM; k0 += 32) {
        __syncthreads();
        // A tile: 128x32 fp32 -> bf16 (4 float4 per thread, packed 8B stores)
        for (int it = 0; it < 4; ++it) {
            int idx = (tid + it * 256) * 4;          // 0..4095
            int r = idx >> 5, c = idx & 31;
            int gr = mbase + r;
            float4 f = make_float4(0.f, 0.f, 0.f, 0.f);
            if (gr < MROWS) {
                f = *(const float4*)(x + (size_t)gr * CDIM + k0 + c);
                __builtin_prefetch(x + (size_t)gr * CDIM + k0 + c + 32, 0, 1);
            }
            *(uint2*)&As[r][c] = pack4bf(f.x, f.y, f.z, f.w);
        }
        // B tile: 32x64 fp32, stored transposed BsT[col][k]
        for (int it = 0; it < 2; ++it) {
            int idx = (tid + it * 256) * 4;          // 0..2047
            int r = idx >> 6, c = idx & 63;
            float4 f = *(const float4*)(Wqkv + (size_t)(k0 + r) * 1536 + nbase + c);
            BsT[c + 0][r] = (bf16)f.x; BsT[c + 1][r] = (bf16)f.y;
            BsT[c + 2][r] = (bf16)f.z; BsT[c + 3][r] = (bf16)f.w;
        }
        __syncthreads();

        // A fragment: contiguous 32B of row `arow`
        const int arow = wave * 16 + l16;
        v16bf afr = load16bf(&As[arow][half * 16]);
        for (int nt = 0; nt < 4; ++nt) {
            const int col = nt * 16 + l16;
            v16bf bfr = load16bf(&BsT[col][half * 16]);   // contiguous 32B
            acc[nt] = wmma_bf16f32(afr, bfr, acc[nt]);
        }
    }

    // Epilogue: + bias, scale q, scatter
    for (int nt = 0; nt < 4; ++nt) {
        const int col   = nbase + nt * 16 + l16;     // 0..1535
        const int which = col / CDIM;                // 0=q 1=k 2=v
        const int c512  = col % CDIM;
        const int h     = c512 >> 5;
        const int d     = c512 & 31;
        const float bias = bqkv[col];
        const float sc   = (which == 0) ? SCALE_Q : 1.0f;
        for (int j = 0; j < 8; ++j) {
            const int grow = mbase + wave * 16 + half * 8 + j;
            if (grow >= MROWS) continue;
            const int b = grow / NTOK, n = grow % NTOK;
            const bf16 val = (bf16)((acc[nt][j] + bias) * sc);
            if (which == 0) {
                qws[(((size_t)b * NHEAD + h) * NTOK + n) * HDIM + d] = val;
            } else if (which == 1) {
                kws[(((size_t)b * NHEAD + h) * NTOK + n) * HDIM + d] = val;
            } else {
                vtws[(((size_t)b * NHEAD + h) * HDIM + d) * KPAD + n] = val;
            }
        }
    }
}

// ---------------------------------------------------------------------------
// On-the-fly Swin relative-position bias (4 quadrants of the 740x740 matrix)
// ---------------------------------------------------------------------------
__device__ __forceinline__ float rel_bias(
    int h, int qg, int kg,
    const float* __restrict__ bt_target,        // [1849][16]
    const float* __restrict__ bt_temp,          // [961][16]
    const float* __restrict__ temp_target_tab,  // [16][484][1]
    const float* __restrict__ target_temp_tab,  // [16][256][1]
    const float* __restrict__ temp_target_lin,  // [16][1][256]
    const float* __restrict__ target_temp_lin)  // [16][1][484]
{
    if (qg < TEMP_N) {
        if (kg < TEMP_N) {
            const int r0 = (qg >> 4) - (kg >> 4);
            const int r1 = (qg & 15) - (kg & 15);
            const int idx = (r0 + TEMP_L - 1) * (2 * TEMP_L - 1) + (r1 + TEMP_L - 1);
            return bt_temp[idx * NHEAD + h];
        }
        return target_temp_tab[h * TEMP_N + qg] + target_temp_lin[h * TARGET_N + (kg - TEMP_N)];
    }
    if (kg < TEMP_N) {
        return temp_target_tab[h * TARGET_N + (qg - TEMP_N)] + temp_target_lin[h * TEMP_N + kg];
    }
    const int qi = qg - TEMP_N, ki = kg - TEMP_N;
    const int r0 = qi / TARGET_L - ki / TARGET_L;
    const int r1 = qi % TARGET_L - ki % TARGET_L;
    const int idx = (r0 + TARGET_L - 1) * (2 * TARGET_L - 1) + (r1 + TARGET_L - 1);
    return bt_target[idx * NHEAD + h];
}

// ---------------------------------------------------------------------------
// Kernel 2: attention for one (b, h, 64-query block).
// 128 threads = 4 waves, each wave owns 16 query rows.
// ---------------------------------------------------------------------------
__global__ __launch_bounds__(128)
void attn_kernel(const bf16* __restrict__ qws, const bf16* __restrict__ kws,
                 const bf16* __restrict__ vtws,  // [B,H,hd,KPAD]
                 const float* __restrict__ bt_target, const float* __restrict__ bt_temp,
                 const float* __restrict__ temp_target_tab, const float* __restrict__ target_temp_tab,
                 const float* __restrict__ temp_target_lin, const float* __restrict__ target_temp_lin,
                 bf16* __restrict__ attn_out)   // [MROWS][512] bf16
{
    extern __shared__ float smem[];
    float* S      = smem;                              // [64][768] f32
    bf16*  Pbf    = (bf16*)(smem + S_FLOATS);          // [64][768] bf16
    float* rowsum = (float*)((char*)smem + S_FLOATS * 4 + QBLK * KPAD * 2);  // [64]

    const int tid  = threadIdx.x;
    const int wave = tid >> 5;
    const int lane = tid & 31;
    const int half = lane >> 4;
    const int l16  = lane & 15;

    const int qb = blockIdx.x % 12;       // 12 * 64 = 768 >= 740
    const int bh = blockIdx.x / 12;
    const int b  = bh >> 4;
    const int h  = bh & 15;
    const int qbase = qb * QBLK;

    const size_t head_base = ((size_t)b * NHEAD + h) * NTOK;         // q/k rows
    const size_t vhead     = ((size_t)b * NHEAD + h) * HDIM;         // vt rows

    // ---- q fragment for this wave's 16 rows (A operand, 16x32 bf16) ----
    const int qrow_local = wave * 16 + l16;
    const int qg_row = qbase + qrow_local;
    const int qclamp = (qg_row < NTOK) ? qg_row : (NTOK - 1);
    const bf16* qp = qws + (head_base + qclamp) * HDIM;
    const v16bf aq = load16bf(qp + half * 16);

    // ---- Pass 1: score panel S = q k^T + bias ----
    for (int kt = 0; kt < KPAD / 16; ++kt) {
        const int key = kt * 16 + l16;
        v16bf bk;
        if (key < NTOK) {
            bk = load16bf(kws + (head_base + key) * HDIM + half * 16);  // contiguous 32B
        } else {
            for (int i = 0; i < 16; ++i) bk[i] = (bf16)0.0f;
        }
        v8f c;
        for (int j = 0; j < 8; ++j) c[j] = 0.0f;
        c = wmma_bf16f32(aq, bk, c);

        for (int j = 0; j < 8; ++j) {
            const int rl = wave * 16 + half * 8 + j;
            const int qg = qbase + rl;
            float s = -1e30f;
            if (key < NTOK && qg < NTOK) {
                s = c[j] + rel_bias(h, qg, key, bt_target, bt_temp,
                                    temp_target_tab, target_temp_tab,
                                    temp_target_lin, target_temp_lin);
            }
            S[(size_t)rl * KPAD + kt * 16 + l16] = s;
        }
    }
    __syncthreads();

    // ---- softmax: thread t owns row t; write exp as bf16 into Pbf ----
    if (tid < QBLK) {
        const float4* r4 = (const float4*)(S + (size_t)tid * KPAD);
        float m = -1e30f;
        for (int c = 0; c < KPAD / 4; ++c) {
            const float4 f = r4[c];
            m = fmaxf(m, fmaxf(fmaxf(f.x, f.y), fmaxf(f.z, f.w)));
        }
        float l = 0.0f;
        uint2* p2 = (uint2*)(Pbf + (size_t)tid * KPAD);
        for (int c = 0; c < KPAD / 4; ++c) {
            const float4 f = r4[c];
            const float e0 = __expf(f.x - m), e1 = __expf(f.y - m);
            const float e2 = __expf(f.z - m), e3 = __expf(f.w - m);
            l += (e0 + e1) + (e2 + e3);
            p2[c] = pack4bf(e0, e1, e2, e3);
        }
        rowsum[tid] = (l > 0.0f) ? l : 1.0f;
    }
    __syncthreads();

    // ---- Pass 2: O = P @ V, hd split into two 16-wide WMMA tiles ----
    v8f o[2];
    for (int t = 0; t < 2; ++t)
        for (int j = 0; j < 8; ++j) o[t][j] = 0.0f;

    const bf16* prow = Pbf + (size_t)(wave * 16 + l16) * KPAD;
    for (int kb = 0; kb < KPAD / 32; ++kb) {
        // P fragment (A operand): contiguous 32B LDS
        const v16bf pa = load16bf(prow + kb * 32 + half * 16);
        const int keyb = kb * 32 + half * 16;
        for (int t = 0; t < 2; ++t) {
            // V fragment (B operand): contiguous in key thanks to vt layout
            const v16bf bv = load16bf(vtws + (vhead + t * 16 + l16) * KPAD + keyb);
            o[t] = wmma_bf16f32(pa, bv, o[t]);
        }
    }

    // ---- store: attn_out[b*740+q][h*32 + d], normalized by row sum ----
    for (int t = 0; t < 2; ++t) {
        for (int j = 0; j < 8; ++j) {
            const int rl = wave * 16 + half * 8 + j;
            const int qg = qbase + rl;
            if (qg < NTOK) {
                const float val = o[t][j] / rowsum[rl];
                attn_out[((size_t)b * NTOK + qg) * CDIM + h * HDIM + t * 16 + l16] = (bf16)val;
            }
        }
    }
}

// ---------------------------------------------------------------------------
// Kernel 3: out = attn_out @ W_proj + b_proj  (bf16 WMMA, f32 out).
// A tile (already bf16 in ws) is DMA'd by the Tensor Data Mover while the
// waves convert/transpose the fp32 W tile - TDM overlap with real work.
// ---------------------------------------------------------------------------
__global__ __launch_bounds__(256)
void proj_kernel(const bf16* __restrict__ ao,     // [MROWS][512] bf16
                 const float* __restrict__ Wproj, // [512][512]
                 const float* __restrict__ bproj, // [512]
                 float* __restrict__ out)         // [MROWS][512]
{
    __shared__ __align__(16) bf16 As[128][32];
    __shared__ __align__(16) bf16 BsT[64][32];

    const int tid  = threadIdx.x;
    const int wave = tid >> 5;
    const int lane = tid & 31;
    const int half = lane >> 4;
    const int l16  = lane & 15;
    const int mbase = blockIdx.x * 128;
    const int nbase = blockIdx.y * 64;

    const unsigned int ldsA = (unsigned int)(uintptr_t)(&As[0][0]);  // LDS byte offset
    const unsigned int rows_avail = (unsigned int)(MROWS - mbase);   // HW zero-fills past this

    v8f acc[4];
    for (int nt = 0; nt < 4; ++nt)
        for (int j = 0; j < 8; ++j) acc[nt][j] = 0.0f;

    for (int k0 = 0; k0 < CDIM; k0 += 32) {
        __syncthreads();
        // A tile via TDM: 128 rows x 32 bf16, row stride 512 elements.
        // One wave issues (TDM ignores EXEC -> gate per-wave), others fill B.
        if (wave == 0) {
            tdm_load_tile_bf16(ldsA, ao + (size_t)mbase * CDIM + k0,
                               /*tile_d0=*/32, /*tile_d1=*/128,
                               /*stride0=*/CDIM, rows_avail);
        }
        // B tile: 32x64 fp32 -> bf16, stored transposed BsT[col][k]
        for (int it = 0; it < 2; ++it) {
            int idx = (tid + it * 256) * 4;
            int r = idx >> 6, c = idx & 63;
            float4 f = *(const float4*)(Wproj + (size_t)(k0 + r) * CDIM + nbase + c);
            BsT[c + 0][r] = (bf16)f.x; BsT[c + 1][r] = (bf16)f.y;
            BsT[c + 2][r] = (bf16)f.z; BsT[c + 3][r] = (bf16)f.w;
        }
        if (wave == 0) {
            __builtin_amdgcn_s_wait_tensorcnt(0);
        }
        __syncthreads();

        const int arow = wave * 16 + l16;
        v16bf afr = load16bf(&As[arow][half * 16]);
        for (int nt = 0; nt < 4; ++nt) {
            const int col = nt * 16 + l16;
            v16bf bfr = load16bf(&BsT[col][half * 16]);
            acc[nt] = wmma_bf16f32(afr, bfr, acc[nt]);
        }
    }

    for (int nt = 0; nt < 4; ++nt) {
        const int col = nbase + nt * 16 + l16;
        const float bias = bproj[col];
        for (int j = 0; j < 8; ++j) {
            const int grow = mbase + wave * 16 + half * 8 + j;
            if (grow < MROWS)
                out[(size_t)grow * CDIM + col] = acc[nt][j] + bias;
        }
    }
}

// ---------------------------------------------------------------------------
// Host launcher
// ---------------------------------------------------------------------------
extern "C" void kernel_launch(void* const* d_in, const int* in_sizes, int n_in,
                              void* d_out, int out_size, void* d_ws, size_t ws_size,
                              hipStream_t stream)
{
    const float* x                = (const float*)d_in[0];
    const float* Wqkv             = (const float*)d_in[1];
    const float* bqkv             = (const float*)d_in[2];
    const float* Wproj            = (const float*)d_in[3];
    const float* bproj            = (const float*)d_in[4];
    const float* bt_target        = (const float*)d_in[5];
    const float* bt_temp          = (const float*)d_in[6];
    const float* temp_target_tab  = (const float*)d_in[7];
    const float* target_temp_tab  = (const float*)d_in[8];
    const float* temp_target_lin  = (const float*)d_in[9];
    const float* target_temp_lin  = (const float*)d_in[10];
    float* out = (float*)d_out;

    // workspace: q | k ([B,H,N,hd] bf16) | vt ([B,H,hd,KPAD] bf16) | attn_out
    char* ws = (char*)d_ws;
    const size_t per_qk = (size_t)BATCH * NHEAD * NTOK * HDIM * sizeof(bf16);   // 12,124,160 B
    const size_t per_vt = (size_t)BATCH * NHEAD * HDIM * KPAD * sizeof(bf16);   // 12,582,912 B
    bf16* qws  = (bf16*)(ws);
    bf16* kws  = (bf16*)(ws + per_qk);
    bf16* vtws = (bf16*)(ws + 2 * per_qk);
    bf16* aows = (bf16*)(ws + 2 * per_qk + per_vt);

    (void)hipFuncSetAttribute((const void*)attn_kernel,
                              hipFuncAttributeMaxDynamicSharedMemorySize, ATTN_LDS);

    // 1) QKV projection GEMM (M=11840, N=1536, K=512)
    qkv_kernel<<<dim3((MROWS + 127) / 128, 1536 / 64), 256, 0, stream>>>(
        x, Wqkv, bqkv, qws, kws, vtws);

    // 2) attention: B*H*(ceil(740/64)=12) workgroups
    attn_kernel<<<dim3(BATCH * NHEAD * 12), 128, ATTN_LDS, stream>>>(
        qws, kws, vtws, bt_target, bt_temp,
        temp_target_tab, target_temp_tab, temp_target_lin, target_temp_lin,
        aows);

    // 3) output projection GEMM (M=11840, N=512, K=512)
    proj_kernel<<<dim3((MROWS + 127) / 128, CDIM / 64), 256, 0, stream>>>(
        aows, Wproj, bproj, out);
}
